// Head_9010841387309
// MI455X (gfx1250) — matrix operand order repaired and checked by reading, not compiled, not run on hardware
//
#include <hip/hip_runtime.h>
#include <hip/hip_bf16.h>

#define BATCH 4
#define SEQ   4096
#define EMB   1024
#define HEAD  64

typedef __attribute__((ext_vector_type(16))) _Float16 v16h;
typedef __attribute__((ext_vector_type(8)))  _Float16 v8h;
typedef __attribute__((ext_vector_type(8)))  float    v8f;

union AFrag { v16h v; v8h h[2]; };

__device__ __forceinline__ v8f wmma_f16(v16h a, v16h b, v8f c) {
  return __builtin_amdgcn_wmma_f32_16x16x32_f16(false, a, false, b, (short)0, c, false, false);
}

__device__ __forceinline__ float fast_exp2(float x) {
#if __has_builtin(__builtin_amdgcn_exp2f)
  return __builtin_amdgcn_exp2f(x);
#else
  return exp2f(x);
#endif
}

// ---------------------------------------------------------------------------
// Kernel 0: W [E][H] f32  ->  Wt [m][H][E] f16  (transposed so B-fragments of
// the projection GEMM are contiguous 32B loads per lane)
// ---------------------------------------------------------------------------
__global__ void wconv_kernel(const float* __restrict__ Wq,
                             const float* __restrict__ Wk,
                             const float* __restrict__ Wv,
                             _Float16* __restrict__ Wt) {
  int idx = blockIdx.x * 256 + threadIdx.x;
  if (idx >= 3 * HEAD * EMB) return;
  int m = idx >> 16;          // 64*1024 per matrix
  int h = (idx >> 10) & 63;
  int e = idx & 1023;
  const float* W = (m == 0) ? Wq : (m == 1) ? Wk : Wv;
  Wt[idx] = (_Float16)W[e * HEAD + h];
}

// ---------------------------------------------------------------------------
// Kernel 1: QKV projection.  12 waves/WG: wave = (matrix m, h-tile ht).
// 16-row x tile staged f32->f16 in LDS once, shared by all 12 waves.
//   Qh [b][t][h] f16, pre-scaled by 0.125*log2(e)  (softmax in exp2 domain)
//   Kh [b][t][h] f16
//   Vt [b][h][t] f16  (transposed for contiguous B-fragments in P@V)
// ---------------------------------------------------------------------------
#define LDS_STRIDE (EMB + 8)   // halfs; pad to spread LDS banks

__global__ __launch_bounds__(384) void qkv_proj_kernel(
    const float* __restrict__ x, const _Float16* __restrict__ Wt,
    _Float16* __restrict__ Qh, _Float16* __restrict__ Kh,
    _Float16* __restrict__ Vt) {
  __shared__ _Float16 lds[16 * LDS_STRIDE];

  const int bid = blockIdx.x;            // 0..1023
  const int b   = bid >> 8;
  const int tb  = (bid & 255) << 4;
  const int tid = threadIdx.x;

  // --- stage x[16][1024] f32 -> f16 into LDS (float2-vectorized) ---
  const float2* x2 = reinterpret_cast<const float2*>(
      x + (size_t)(b * SEQ + tb) * EMB);
  for (int i = tid; i < 16 * (EMB / 2); i += 384) {
    int row = i >> 9;
    int c   = i & 511;
    float2 v = x2[row * (EMB / 2) + c];
    lds[row * LDS_STRIDE + 2 * c]     = (_Float16)v.x;
    lds[row * LDS_STRIDE + 2 * c + 1] = (_Float16)v.y;
  }
  __syncthreads();

  const int wave = tid >> 5, lane = tid & 31;
  const int m  = wave >> 2;        // 0=Q 1=K 2=V
  const int ht = wave & 3;         // h-tile (16 cols)
  const int l15 = lane & 15, hi = (lane >> 4) & 1;

  // B-fragment source: Wt row for column h = ht*16 + l15, e-chunk per half-wave
  const _Float16* wrow = Wt + (size_t)m * (HEAD * EMB)
                            + (size_t)(ht * 16 + l15) * EMB + hi * 16;
  // A-fragment source: LDS row t = l15, contiguous 16-e chunk per half-wave
  const _Float16* arow = &lds[l15 * LDS_STRIDE + hi * 16];

  v8f acc = {};
  #pragma unroll 8
  for (int eb = 0; eb < EMB; eb += 32) {
    AFrag a;
    a.h[0] = *reinterpret_cast<const v8h*>(arow + eb);        // e: +0..7
    a.h[1] = *reinterpret_cast<const v8h*>(arow + eb + 8);    // e: +8..15
    v16h bf = *reinterpret_cast<const v16h*>(wrow + eb);
    acc = wmma_f16(a.v, bf, acc);
  }

  // C/D layout: VGPR r -> row (tb + hi*8 + r), col ht*16 + l15
  const int trow0 = tb + hi * 8;
  const int hcol  = ht * 16 + l15;
  if (m == 0) {
    const float s = 0.125f * 1.44269504f;  // 1/sqrt(64) * log2(e)
    #pragma unroll
    for (int r = 0; r < 8; ++r)
      Qh[((size_t)(b * SEQ) + trow0 + r) * HEAD + hcol] = (_Float16)(acc[r] * s);
  } else if (m == 1) {
    #pragma unroll
    for (int r = 0; r < 8; ++r)
      Kh[((size_t)(b * SEQ) + trow0 + r) * HEAD + hcol] = (_Float16)acc[r];
  } else {
    #pragma unroll
    for (int r = 0; r < 8; ++r)
      Vt[((size_t)b * HEAD + hcol) * SEQ + trow0 + r] = (_Float16)acc[r];
  }
}

// ---------------------------------------------------------------------------
// Kernel 2: fused flash attention.
//   - Each wave owns TWO 16-query tiles (32 queries): K/V fragments loaded
//     once feed 2x the WMMA work -> halves L2 traffic (the binding roofline).
//   - 4 waves per block split the key dimension (1024 keys each) with private
//     online-softmax state -> 2048 waves for latency hiding.
//   - S^T = K @ Q^T so softmax stats are per-lane; S^T C-layout packs
//     lane-locally into P's A-fragment; O += P @ V.
//   - Partial (m,l,O) merged via log-sum-exp through LDS, coalesced stores.
// ---------------------------------------------------------------------------
#define KV_SPLIT 4
#define KEYS_PER_WAVE (SEQ / KV_SPLIT)
#define QTILE 32                       // queries per wave (2 WMMA tiles)

__global__ __launch_bounds__(32 * KV_SPLIT) void flash_attn_kernel(
    const _Float16* __restrict__ Qh, const _Float16* __restrict__ Kh,
    const _Float16* __restrict__ Vt, float* __restrict__ out) {
  __shared__ float sO[KV_SPLIT * QTILE * HEAD];   // per-wave partial O, 32 KB
  __shared__ float sM[KV_SPLIT * QTILE];          // per-wave row max
  __shared__ float sL[KV_SPLIT * QTILE];          // per-wave row sum

  const int b    = blockIdx.x >> 7;               // SEQ/QTILE = 128 blocks/batch
  const int qb   = (blockIdx.x & 127) << 5;
  const int tid  = threadIdx.x;
  const int w    = tid >> 5;                      // wave = key slice
  const int lane = tid & 31;
  const int l15 = lane & 15, hi = (lane >> 4) & 1;

  // Q^T B-fragments: lane = query column, h rows per half-wave; two q-tiles
  const _Float16* qrow = Qh + ((size_t)(b * SEQ) + qb + l15) * HEAD + hi * 16;
  const v16h bqA0 = *reinterpret_cast<const v16h*>(qrow);             // qtile A, h 0..31
  const v16h bqA1 = *reinterpret_cast<const v16h*>(qrow + 32);        // qtile A, h 32..63
  const v16h bqB0 = *reinterpret_cast<const v16h*>(qrow + 16 * HEAD); // qtile B
  const v16h bqB1 = *reinterpret_cast<const v16h*>(qrow + 16 * HEAD + 32);

  const _Float16* kbase = Kh + (size_t)(b * SEQ) * HEAD;
  const _Float16* vbase = Vt + (size_t)(b * HEAD) * SEQ;

  v8f oA0 = {}, oA1 = {}, oA2 = {}, oA3 = {};
  v8f oB0 = {}, oB1 = {}, oB2 = {}, oB3 = {};
  float mA = -__builtin_huge_valf(), lA = 0.0f;
  float mB = -__builtin_huge_valf(), lB = 0.0f;

  const int kend = (w + 1) * KEYS_PER_WAVE;
  for (int kb = w * KEYS_PER_WAVE; kb < kend; kb += 32) {
    // A-fragments of K: row = key, contiguous h-chunks per half-wave (h = 16*hi + s), same order as Q^T
    const _Float16* krow0 = kbase + (size_t)(kb + l15) * HEAD + hi * 16;
    const _Float16* krow1 = krow0 + 16 * HEAD;

    AFrag a00, a01, a10, a11;
    a00.h[0] = *reinterpret_cast<const v8h*>(krow0);
    a00.h[1] = *reinterpret_cast<const v8h*>(krow0 + 8);
    a01.h[0] = *reinterpret_cast<const v8h*>(krow0 + 32);
    a01.h[1] = *reinterpret_cast<const v8h*>(krow0 + 40);
    a10.h[0] = *reinterpret_cast<const v8h*>(krow1);
    a10.h[1] = *reinterpret_cast<const v8h*>(krow1 + 8);
    a11.h[0] = *reinterpret_cast<const v8h*>(krow1 + 32);
    a11.h[1] = *reinterpret_cast<const v8h*>(krow1 + 40);

    // V B-fragments hoisted: independent of S, loads overlap the softmax chain
    const _Float16* vcol = vbase + (size_t)l15 * SEQ + kb + hi * 8;
    const v16h bv0 = __builtin_shufflevector(*reinterpret_cast<const v8h*>(vcol), *reinterpret_cast<const v8h*>(vcol + 16), 0, 1, 2, 3, 4, 5, 6, 7, 8, 9, 10, 11, 12, 13, 14, 15);
    const v16h bv1 = __builtin_shufflevector(*reinterpret_cast<const v8h*>(vcol + 16 * SEQ), *reinterpret_cast<const v8h*>(vcol + 16 * SEQ + 16), 0, 1, 2, 3, 4, 5, 6, 7, 8, 9, 10, 11, 12, 13, 14, 15);
    const v16h bv2 = __builtin_shufflevector(*reinterpret_cast<const v8h*>(vcol + 32 * SEQ), *reinterpret_cast<const v8h*>(vcol + 32 * SEQ + 16), 0, 1, 2, 3, 4, 5, 6, 7, 8, 9, 10, 11, 12, 13, 14, 15);
    const v16h bv3 = __builtin_shufflevector(*reinterpret_cast<const v8h*>(vcol + 48 * SEQ), *reinterpret_cast<const v8h*>(vcol + 48 * SEQ + 16), 0, 1, 2, 3, 4, 5, 6, 7, 8, 9, 10, 11, 12, 13, 14, 15);

    // scores for both q-tiles (K fragments amortized over 2x WMMA work)
    v8f saA = {}, sbA = {}, saB = {}, sbB = {};
    saA = wmma_f16(a00.v, bqA0, saA);  saA = wmma_f16(a01.v, bqA1, saA);
    sbA = wmma_f16(a10.v, bqA0, sbA);  sbA = wmma_f16(a11.v, bqA1, sbA);
    saB = wmma_f16(a00.v, bqB0, saB);  saB = wmma_f16(a01.v, bqB1, saB);
    sbB = wmma_f16(a10.v, bqB0, sbB);  sbB = wmma_f16(a11.v, bqB1, sbB);

    // --- online softmax, q-tile A (exp2 domain; scale folded into Q) ---
    {
      float tmax = saA[0];
      #pragma unroll
      for (int j = 1; j < 8; ++j) tmax = fmaxf(tmax, saA[j]);
      #pragma unroll
      for (int j = 0; j < 8; ++j) tmax = fmaxf(tmax, sbA[j]);
      tmax = fmaxf(tmax, __shfl_xor(tmax, 16, 32));
      float mnew  = fmaxf(mA, tmax);
      float alpha = fast_exp2(mA - mnew);
      mA = mnew;
      float rs = 0.0f;
      v16h ap;
      #pragma unroll
      for (int j = 0; j < 8; ++j) { float p = fast_exp2(saA[j] - mnew); rs += p; ap[j] = (_Float16)p; }
      #pragma unroll
      for (int j = 0; j < 8; ++j) { float p = fast_exp2(sbA[j] - mnew); rs += p; ap[8 + j] = (_Float16)p; }
      rs += __shfl_xor(rs, 16, 32);
      lA = lA * alpha + rs;
      #pragma unroll
      for (int r = 0; r < 8; ++r) {
        float ar = __shfl(alpha, r + hi * 24, 32);
        oA0[r] *= ar; oA1[r] *= ar; oA2[r] *= ar; oA3[r] *= ar;
      }
      oA0 = wmma_f16(ap, bv0, oA0);
      oA1 = wmma_f16(ap, bv1, oA1);
      oA2 = wmma_f16(ap, bv2, oA2);
      oA3 = wmma_f16(ap, bv3, oA3);
    }
    // --- online softmax, q-tile B ---
    {
      float tmax = saB[0];
      #pragma unroll
      for (int j = 1; j < 8; ++j) tmax = fmaxf(tmax, saB[j]);
      #pragma unroll
      for (int j = 0; j < 8; ++j) tmax = fmaxf(tmax, sbB[j]);
      tmax = fmaxf(tmax, __shfl_xor(tmax, 16, 32));
      float mnew  = fmaxf(mB, tmax);
      float alpha = fast_exp2(mB - mnew);
      mB = mnew;
      float rs = 0.0f;
      v16h ap;
      #pragma unroll
      for (int j = 0; j < 8; ++j) { float p = fast_exp2(saB[j] - mnew); rs += p; ap[j] = (_Float16)p; }
      #pragma unroll
      for (int j = 0; j < 8; ++j) { float p = fast_exp2(sbB[j] - mnew); rs += p; ap[8 + j] = (_Float16)p; }
      rs += __shfl_xor(rs, 16, 32);
      lB = lB * alpha + rs;
      #pragma unroll
      for (int r = 0; r < 8; ++r) {
        float ar = __shfl(alpha, r + hi * 24, 32);
        oB0[r] *= ar; oB1[r] *= ar; oB2[r] *= ar; oB3[r] *= ar;
      }
      oB0 = wmma_f16(ap, bv0, oB0);
      oB1 = wmma_f16(ap, bv1, oB1);
      oB2 = wmma_f16(ap, bv2, oB2);
      oB3 = wmma_f16(ap, bv3, oB3);
    }
  }

  // --- publish partial state (q index within block: A = 0..15, B = 16..31) ---
  if (!hi) {
    sM[w * QTILE + l15]      = mA;  sL[w * QTILE + l15]      = lA;
    sM[w * QTILE + 16 + l15] = mB;  sL[w * QTILE + 16 + l15] = lB;
  }
  #pragma unroll
  for (int r = 0; r < 8; ++r) {
    int q = hi * 8 + r;
    float* dA = &sO[(w * QTILE + q) * HEAD + l15];
    dA[0] = oA0[r]; dA[16] = oA1[r]; dA[32] = oA2[r]; dA[48] = oA3[r];
    float* dB = &sO[(w * QTILE + 16 + q) * HEAD + l15];
    dB[0] = oB0[r]; dB[16] = oB1[r]; dB[32] = oB2[r]; dB[48] = oB3[r];
  }
  __syncthreads();

  // --- log-sum-exp merge of the 4 key slices + coalesced store ---
  // thread t -> query row q = t>>2 (0..31), h-chunk hb = (t&3)*16
  {
    const int q  = tid >> 2;
    const int hb = (tid & 3) * 16;
    float m0 = sM[q], m1 = sM[QTILE + q], m2 = sM[2 * QTILE + q], m3 = sM[3 * QTILE + q];
    float mg = fmaxf(fmaxf(m0, m1), fmaxf(m2, m3));
    float f0 = fast_exp2(m0 - mg), f1 = fast_exp2(m1 - mg);
    float f2 = fast_exp2(m2 - mg), f3 = fast_exp2(m3 - mg);
    float lg = sL[q] * f0 + sL[QTILE + q] * f1
             + sL[2 * QTILE + q] * f2 + sL[3 * QTILE + q] * f3;
    float inv = 1.0f / lg;

    const float* r0 = &sO[(0 * QTILE + q) * HEAD + hb];
    const float* r1 = &sO[(1 * QTILE + q) * HEAD + hb];
    const float* r2 = &sO[(2 * QTILE + q) * HEAD + hb];
    const float* r3 = &sO[(3 * QTILE + q) * HEAD + hb];
    float* orow = out + ((size_t)(b * SEQ) + qb + q) * HEAD + hb;
    #pragma unroll
    for (int j = 0; j < 16; ++j)
      orow[j] = (r0[j] * f0 + r1[j] * f1 + r2[j] * f2 + r3[j] * f3) * inv;
  }
}

// ---------------------------------------------------------------------------
extern "C" void kernel_launch(void* const* d_in, const int* in_sizes, int n_in,
                              void* d_out, int out_size, void* d_ws, size_t ws_size,
                              hipStream_t stream) {
  const float* x  = (const float*)d_in[0];
  const float* Wq = (const float*)d_in[1];
  const float* Wk = (const float*)d_in[2];
  const float* Wv = (const float*)d_in[3];
  float* out = (float*)d_out;

  char* ws = (char*)d_ws;
  _Float16* Wt = (_Float16*)ws;                               // 3*64*1024*2 = 384 KB
  _Float16* Qh = (_Float16*)(ws + 0x60000);                   // 2 MB
  _Float16* Kh = (_Float16*)(ws + 0x60000 + 0x200000);        // 2 MB
  _Float16* Vt = (_Float16*)(ws + 0x60000 + 0x400000);        // 2 MB

  // weights -> f16 transposed
  wconv_kernel<<<(3 * HEAD * EMB + 255) / 256, 256, 0, stream>>>(Wq, Wk, Wv, Wt);
  // QKV projection (WMMA), Q pre-scaled into exp2 domain
  qkv_proj_kernel<<<BATCH * (SEQ / 16), 384, 0, stream>>>(x, Wt, Qh, Kh, Vt);
  // fused flash attention (WMMA, 32 queries/wave, split-K across 4 waves)
  flash_attn_kernel<<<BATCH * (SEQ / QTILE), 32 * KV_SPLIT, 0, stream>>>(Qh, Kh, Vt, out);
}
